// GatedGCNLayer_57526791962767
// MI455X (gfx1250) — matrix operand order repaired and checked by reading, not compiled, not run on hardware
//
#include <hip/hip_runtime.h>
#include <hip/hip_bf16.h>

#define D_   128
#define NN_  256
#define JT   32
#define AS_  136   // bf16 LDS row stride (elements) -> 272B, 16B-aligned rows
#define GS_  132   // f32 LDS row stride for gate_input -> 528B, 16B-aligned rows
#define EPS_ 1e-5f

typedef __attribute__((ext_vector_type(16))) __bf16       v16bf;
typedef __attribute__((ext_vector_type(8)))  float        v8f;
typedef __attribute__((ext_vector_type(4)))  float        f32x4;
typedef __attribute__((ext_vector_type(4)))  unsigned int u32x4;

union Frag16 {
    v16bf v;
    u32x4 q[2];
    unsigned int u[8];
};

// pack two fp32 -> two bf16 in one uint (low = a, high = b).
// round-half-up (+0x8000) then v_perm_b32 grabs the two high halves:
// sel 0x07060302: dst.b0=a.b2 dst.b1=a.b3 dst.b2=b.b2 dst.b3=b.b3
static __device__ __forceinline__ unsigned int cvt2(float a, float b) {
    unsigned int ua = __float_as_uint(a) + 0x8000u;
    unsigned int ub = __float_as_uint(b) + 0x8000u;
    return __builtin_amdgcn_perm(ub, ua, 0x07060302u);
}

// convert fp32 -> bf16 fragment: elements 0..7 from p[0..7], elements 8..15 from p[16..23]
static __device__ __forceinline__ v16bf cvt16(const float* __restrict__ p) {
    Frag16 f;
#pragma unroll
    for (int i = 0; i < 8; ++i) f.u[i] = cvt2(p[2 * i + (i >= 4 ? 8 : 0)], p[2 * i + 1 + (i >= 4 ? 8 : 0)]);
    return f.v;
}

// A-fragment from LDS: elements 0..7 at p, elements 8..15 at p+16 (K chunk, K chunk+16)
static __device__ __forceinline__ v16bf ldsA(const unsigned short* __restrict__ p) {
    Frag16 f;
    f.q[0] = *(const u32x4*)(p);
    f.q[1] = *(const u32x4*)(p + 16);
    return f.v;
}

// ---------------------------------------------------------------------------
// prep: Ah/Bh/Uh/Vh = h @ W^T + b   (rows = B*N = 1024, D=128) via bf16 WMMA
// ---------------------------------------------------------------------------
__global__ __launch_bounds__(256) void prep_kernel(
    const float* __restrict__ h,
    const float* __restrict__ WA, const float* __restrict__ bA,
    const float* __restrict__ WB, const float* __restrict__ bB,
    const float* __restrict__ WU, const float* __restrict__ bU,
    const float* __restrict__ WV, const float* __restrict__ bV,
    float* __restrict__ Ah, float* __restrict__ Bh,
    float* __restrict__ Uh, float* __restrict__ Vh)
{
    __shared__ __align__(16) unsigned short sH[JT * AS_];

    const int t    = threadIdx.x;
    const int lane = t & 31;
    const int wave = t >> 5;
    const int row0 = blockIdx.x * JT;

    // stage 32 rows of h -> bf16 LDS (thread owns row jj, cols d0..d0+15)
    {
        const int jj = t >> 3;
        const int d0 = (t & 7) * 16;
        const float* hr = h + (size_t)(row0 + jj) * D_ + d0;
        float hv[16];
#pragma unroll
        for (int q4 = 0; q4 < 4; ++q4) {
            f32x4 v = *(const f32x4*)(hr + q4 * 4);
            hv[q4 * 4 + 0] = v.x; hv[q4 * 4 + 1] = v.y;
            hv[q4 * 4 + 2] = v.z; hv[q4 * 4 + 3] = v.w;
        }
        Frag16 pk;
#pragma unroll
        for (int i = 0; i < 8; ++i) pk.u[i] = cvt2(hv[2 * i], hv[2 * i + 1]);
        *(u32x4*)&sH[jj * AS_ + d0]     = pk.q[0];
        *(u32x4*)&sH[jj * AS_ + d0 + 8] = pk.q[1];
    }
    __syncthreads();

    const int nt    = wave;               // 8 waves -> 8 N-tiles
    const int ncol  = nt * 16 + (lane & 15);
    const int khalf = lane >> 4;
    const int mrow  = lane & 15;
    const int mh    = khalf * 8;

    const float* Ws[4] = {WA, WB, WU, WV};
    const float* bs[4] = {bA, bB, bU, bV};
    float*       Os[4] = {Ah, Bh, Uh, Vh};

#pragma unroll
    for (int mat = 0; mat < 4; ++mat) {
        Frag16 Bf[4];
        const float* wrow = Ws[mat] + (size_t)ncol * D_;   // Bmat[k][n] = W[n][k]
#pragma unroll
        for (int kt = 0; kt < 4; ++kt) Bf[kt].v = cvt16(wrow + kt * 32 + khalf * 8);
        const float bias = bs[mat][ncol];

#pragma unroll
        for (int mt = 0; mt < 2; ++mt) {
            v8f acc = {bias, bias, bias, bias, bias, bias, bias, bias};
#pragma unroll
            for (int kt = 0; kt < 4; ++kt) {
                v16bf a = ldsA(&sH[(mt * 16 + mrow) * AS_ + kt * 32 + khalf * 8]);
                acc = __builtin_amdgcn_wmma_f32_16x16x32_bf16(
                    false, a, false, Bf[kt].v, (short)0, acc, false, false);
            }
            float* o = Os[mat] + (size_t)(row0 + mt * 16 + mh) * D_ + ncol;
#pragma unroll
            for (int r = 0; r < 8; ++r) o[(size_t)r * D_] = acc[r];
        }
    }
}

// ---------------------------------------------------------------------------
// main: per (b,i) block — fused C(e) GEMM, gate, LN, agg, e_new and h_new
// ---------------------------------------------------------------------------
__global__ __launch_bounds__(256) void gcn_main(
    const float* __restrict__ h,  const float* __restrict__ e,
    const float* __restrict__ WC, const float* __restrict__ bC,
    const float* __restrict__ gh, const float* __restrict__ bh,
    const float* __restrict__ ge, const float* __restrict__ be,
    const float* __restrict__ Ah, const float* __restrict__ Bh,
    const float* __restrict__ Uh, const float* __restrict__ Vh,
    float* __restrict__ hOut, float* __restrict__ eOut)
{
    __shared__ __align__(16) unsigned short sA[JT * AS_]; // e tile, bf16
    __shared__ __align__(16) float sGI[JT * GS_];          // gate_input tile / agg scratch
    __shared__ float sRed[8];

    const int t    = threadIdx.x;
    const int lane = t & 31;
    const int wave = t >> 5;
    const int bi   = blockIdx.x;        // b*N + i
    const int b    = bi >> 8;

    // WC fragments, held in registers for the whole j-loop
    const int nt    = wave;
    const int ncol  = nt * 16 + (lane & 15);
    const int khalf = lane >> 4;
    const int mrow  = lane & 15;
    const int mh    = khalf * 8;
    Frag16 Bf[4];
    {
        const float* wrow = WC + (size_t)ncol * D_;
#pragma unroll
        for (int kt = 0; kt < 4; ++kt) Bf[kt].v = cvt16(wrow + kt * 32 + khalf * 8);
    }
    // Ah[b,i,ncol] + bC[ncol], folded into WMMA accumulator init
    const float cinit = Ah[(size_t)bi * D_ + ncol] + bC[ncol];

    const int jj = t >> 3;            // row this thread stages + processes
    const int d0 = (t & 7) * 16;      // 16-wide d slice

    // per-thread LN affine slice, kept in registers for the whole kernel
    float geR[16], beR[16];
#pragma unroll
    for (int q4 = 0; q4 < 4; ++q4) {
        f32x4 g4 = *(const f32x4*)(ge + d0 + q4 * 4);
        f32x4 b4 = *(const f32x4*)(be + d0 + q4 * 4);
        geR[q4 * 4 + 0] = g4.x; geR[q4 * 4 + 1] = g4.y;
        geR[q4 * 4 + 2] = g4.z; geR[q4 * 4 + 3] = g4.w;
        beR[q4 * 4 + 0] = b4.x; beR[q4 * 4 + 1] = b4.y;
        beR[q4 * 4 + 2] = b4.z; beR[q4 * 4 + 3] = b4.w;
    }

    float agg[16];
#pragma unroll
    for (int q = 0; q < 16; ++q) agg[q] = 0.f;

    const float* etile = e + (size_t)bi * NN_ * D_;

    for (int j0 = 0; j0 < NN_; j0 += JT) {
        // ---- stage e[b,i,j0+jj, d0..d0+15]: keep fp32 in regs, bf16 -> LDS ----
        float ef[16];
        {
            const float* erow = etile + (size_t)(j0 + jj) * D_ + d0;
#pragma unroll
            for (int q4 = 0; q4 < 4; ++q4) {
                f32x4 v = *(const f32x4*)(erow + q4 * 4);
                ef[q4 * 4 + 0] = v.x; ef[q4 * 4 + 1] = v.y;
                ef[q4 * 4 + 2] = v.z; ef[q4 * 4 + 3] = v.w;
            }
            Frag16 pk;
#pragma unroll
            for (int i = 0; i < 8; ++i) pk.u[i] = cvt2(ef[2 * i], ef[2 * i + 1]);
            *(u32x4*)&sA[jj * AS_ + d0]     = pk.q[0];
            *(u32x4*)&sA[jj * AS_ + d0 + 8] = pk.q[1];
        }
        // prefetch next tile (lowers to global_prefetch_b8)
        if (j0 + JT < NN_) {
            __builtin_prefetch(etile + (size_t)(j0 + JT) * D_ + (size_t)t * 16, 0, 1);
        }
        __syncthreads();

        // ---- WMMA: Ce tile (32 x 128); wave owns N-tile nt, both M-tiles.
        //      Accumulators pre-loaded with Ah[b,i,ncol]+bC[ncol].
        v8f acc0 = {cinit, cinit, cinit, cinit, cinit, cinit, cinit, cinit};
        v8f acc1 = acc0;
#pragma unroll
        for (int kt = 0; kt < 4; ++kt) {
            v16bf a0 = ldsA(&sA[mrow * AS_ + kt * 32 + khalf * 8]);
            v16bf a1 = ldsA(&sA[(16 + mrow) * AS_ + kt * 32 + khalf * 8]);
            acc0 = __builtin_amdgcn_wmma_f32_16x16x32_bf16(
                false, a0, false, Bf[kt].v, (short)0, acc0, false, false);
            acc1 = __builtin_amdgcn_wmma_f32_16x16x32_bf16(
                false, a1, false, Bf[kt].v, (short)0, acc1, false, false);
        }
#pragma unroll
        for (int r = 0; r < 8; ++r) {
            sGI[(mh + r) * GS_ + ncol]      = acc0[r];
            sGI[(16 + mh + r) * GS_ + ncol] = acc1[r];
        }
        __syncthreads();

        // ---- elementwise: gate_input, LN over d, sigmoid gate, agg, e_new ----
        const int j = j0 + jj;
        float gi[16];
        float s1 = 0.f, s2 = 0.f;
        {
            const f32x4* bh4 = (const f32x4*)(Bh + ((size_t)b * NN_ + j) * D_ + d0);
#pragma unroll
            for (int q4 = 0; q4 < 4; ++q4) {
                f32x4 bv = bh4[q4];
                f32x4 gv = *(const f32x4*)&sGI[jj * GS_ + d0 + q4 * 4];
                float g0 = gv.x + bv.x, g1 = gv.y + bv.y;
                float g2 = gv.z + bv.z, g3 = gv.w + bv.w;
                gi[q4 * 4 + 0] = g0; gi[q4 * 4 + 1] = g1;
                gi[q4 * 4 + 2] = g2; gi[q4 * 4 + 3] = g3;
                s1 += g0 + g1 + g2 + g3;
                s2 = fmaf(g0, g0, fmaf(g1, g1, fmaf(g2, g2, fmaf(g3, g3, s2))));
            }
        }
#pragma unroll
        for (int off = 1; off < 8; off <<= 1) {     // 8 lanes share a row
            s1 += __shfl_xor(s1, off, 32);
            s2 += __shfl_xor(s2, off, 32);
        }
        const float mean = s1 * (1.f / 128.f);
        const float var  = s2 * (1.f / 128.f) - mean * mean;
        const float rstd = rsqrtf(var + EPS_);

        {
            const f32x4* v4  = (const f32x4*)(Vh + ((size_t)b * NN_ + j) * D_ + d0);
            f32x4*       eo4 = (f32x4*)(eOut + ((size_t)bi * NN_ + j) * D_ + d0);
#pragma unroll
            for (int q4 = 0; q4 < 4; ++q4) {
                f32x4 vv = v4[q4];
                float  o[4];
#pragma unroll
                for (int u = 0; u < 4; ++u) {
                    int   q  = q4 * 4 + u;
                    float g  = gi[q];
                    float ln = (g - mean) * rstd * geR[q] + beR[q];
                    o[u] = fmaxf(ln, 0.f) + ef[q];
                    float gate = __builtin_amdgcn_rcpf(1.f + __expf(-g));
                    float vq = (u == 0) ? vv.x : (u == 1) ? vv.y : (u == 2) ? vv.z : vv.w;
                    agg[q] = fmaf(gate, vq, agg[q]);
                }
                f32x4 ov;
                ov.x = o[0]; ov.y = o[1]; ov.z = o[2]; ov.w = o[3];
                __builtin_nontemporal_store(ov, eo4 + q4);   // e_new never re-read
            }
        }
        __syncthreads();
    }

    // ---- reduce agg partials across the 32 j-rows, then fused h_new ----
#pragma unroll
    for (int q = 0; q < 16; ++q) sGI[jj * GS_ + d0 + q] = agg[q];
    __syncthreads();

    float x = 0.f;
    if (t < D_) {
        float a = 0.f;
#pragma unroll 8
        for (int r = 0; r < JT; ++r) a += sGI[r * GS_ + t];
        x = Uh[(size_t)bi * D_ + t] + a;
    }
    float s1 = x, s2 = x * x;
#pragma unroll
    for (int off = 1; off < 32; off <<= 1) {
        s1 += __shfl_xor(s1, off, 32);
        s2 += __shfl_xor(s2, off, 32);
    }
    if (t < D_ && lane == 0) { sRed[wave] = s1; sRed[4 + wave] = s2; }
    __syncthreads();
    if (t < D_) {
        float S1 = sRed[0] + sRed[1] + sRed[2] + sRed[3];
        float S2 = sRed[4] + sRed[5] + sRed[6] + sRed[7];
        float mean = S1 * (1.f / 128.f);
        float var  = S2 * (1.f / 128.f) - mean * mean;
        float rstd = rsqrtf(var + EPS_);
        float ln   = (x - mean) * rstd * gh[t] + bh[t];
        hOut[(size_t)bi * D_ + t] = fmaxf(ln, 0.f) + h[(size_t)bi * D_ + t];
    }
}

// ---------------------------------------------------------------------------
extern "C" void kernel_launch(void* const* d_in, const int* in_sizes, int n_in,
                              void* d_out, int out_size, void* d_ws, size_t ws_size,
                              hipStream_t stream) {
    const float* h  = (const float*)d_in[0];
    const float* e  = (const float*)d_in[1];
    const float* WA = (const float*)d_in[2];
    const float* bA = (const float*)d_in[3];
    const float* WB = (const float*)d_in[4];
    const float* bB = (const float*)d_in[5];
    const float* WC = (const float*)d_in[6];
    const float* bC = (const float*)d_in[7];
    const float* WU = (const float*)d_in[8];
    const float* bU = (const float*)d_in[9];
    const float* WV = (const float*)d_in[10];
    const float* bV = (const float*)d_in[11];
    const float* gh = (const float*)d_in[12];
    const float* bh = (const float*)d_in[13];
    const float* ge = (const float*)d_in[14];
    const float* be = (const float*)d_in[15];

    float* ws = (float*)d_ws;
    float* Ah = ws;                 // [1024,128] each
    float* Bh = ws + 131072;
    float* Uh = ws + 262144;
    float* Vh = ws + 393216;

    float* hOut = (float*)d_out;            // [4,256,128]
    float* eOut = hOut + 131072;            // [4,256,256,128]

    prep_kernel<<<32, 256, 0, stream>>>(h, WA, bA, WB, bB, WU, bU, WV, bV,
                                        Ah, Bh, Uh, Vh);
    gcn_main<<<1024, 256, 0, stream>>>(h, e, WC, bC, gh, bh, ge, be,
                                       Ah, Bh, Uh, Vh, hOut, eOut);
}
